// Vsgatmodel_31447750542172
// MI455X (gfx1250) — matrix operand order repaired and checked by reading, not compiled, not run on hardware
//
#include <hip/hip_runtime.h>
#include <hip/hip_bf16.h>

// ---------------- problem constants (match reference) ----------------
#define NN 20000
#define EE 160000
#define PP 10000
#define DD 512
#define SS 128
#define HH 1024
#define CIN (2*DD+SS)   // 1152

typedef __bf16 bf16;
typedef __attribute__((ext_vector_type(16))) __bf16 v16bf;
typedef __attribute__((ext_vector_type(8)))  __bf16 v8bf;
typedef __attribute__((ext_vector_type(8)))  float  v8f;

// ---------------- WMMA helpers ----------------
__device__ inline v8f wmma_bf16(v16bf a, v16bf b, v8f c) {
  return __builtin_amdgcn_wmma_f32_16x16x32_bf16(
      /*neg_a=*/false, a, /*neg_b=*/false, b,
      /*c_mod=*/(short)0, c, /*reuse_a=*/false, /*reuse_b=*/false);
}

// A-fragment (16xK tile, 16-bit): lanes 0-15 row M=lane hold K0..7 / K16..23,
// lanes 16-31 row M=lane-16 hold K8..15 / K24..31 (ISA 7.12.2).
__device__ inline v16bf load_a_frag(const bf16* base, int ldk, int lane, int kt) {
  int row  = lane & 15;
  int half = lane >> 4;
  const bf16* p = base + row * ldk + kt * 32 + half * 8;
  v8bf lo = *(const v8bf*)(p);        // K = half*8 + 0..7        (16B aligned)
  v8bf hi = *(const v8bf*)(p + 16);   // K = 16 + half*8 + 0..7
  v16bf a;
#pragma unroll
  for (int j = 0; j < 8; ++j) { a[j] = lo[j]; a[8 + j] = hi[j]; }
  return a;
}

__device__ inline unsigned enc_f32(float f) {
  unsigned u = __float_as_uint(f);
  return (u & 0x80000000u) ? ~u : (u | 0x80000000u);
}
__device__ inline float dec_f32(unsigned u) {
  u = (u & 0x80000000u) ? (u & 0x7fffffffu) : ~u;
  return __uint_as_float(u);
}

// 4-N-tile register-blocked GEMM panel: D[16 x 64] += A[16 x K] * B[K x 64].
// xs: LDS activations (ldk halves per row). Wf: frag-swizzled weights,
// tile (nt,kt) at (nt*KT + kt)*512 halves, lane's 16 halves contiguous.
// acc[u] preloaded with bias; results left in acc.
template <int KT>
__device__ inline void gemm_panel4(const bf16* xs, int ldk, int lane,
                                   const bf16* Wf, int ntbase, v8f acc[4]) {
  const bf16* bw[4];
#pragma unroll
  for (int u = 0; u < 4; ++u)
    bw[u] = Wf + ((size_t)(ntbase + u) * KT) * 512 + lane * 16;
  for (int kt = 0; kt < KT; ++kt) {
    v16bf a = load_a_frag(xs, ldk, lane, kt);
#pragma unroll
    for (int u = 0; u < 4; ++u) {
      v16bf b = *(const v16bf*)(bw[u] + (size_t)kt * 512);
      acc[u] = wmma_bf16(a, b, acc[u]);
    }
  }
}

// ---------------- weight prep: f32 row-major [K][Nn] -> bf16 frag order ----------------
__global__ __launch_bounds__(256) void k_prep_weight(const float* __restrict__ W,
                                                     bf16* __restrict__ out,
                                                     int K, int Nn) {
  int total = (K / 32) * (Nn / 16) * 512;
  int i = blockIdx.x * 256 + threadIdx.x;
  if (i >= total) return;
  int j    = i & 15;
  int lane = (i >> 4) & 31;
  int ft   = i >> 9;
  int KT   = K / 32;
  int kt   = ft % KT;
  int nt   = ft / KT;
  int k = kt * 32 + (lane >> 4) * 8 + (j & 7) + (j >> 3) * 16;   // B[k][n]: K = half*8 + 0..7 / 16 + half*8 + 0..7
  int n = nt * 16 + (lane & 15);
  out[i] = (bf16)W[(size_t)k * Nn + n];
}

// ---------------- edge MLP: e_f = relu(x@We1+be1)@We2+be2, x gathered ----------------
__global__ __launch_bounds__(256) void k_edge_mlp(
    const float* __restrict__ n_f, const float* __restrict__ s_f,
    const bf16* __restrict__ We1f, const float* __restrict__ be1,
    const bf16* __restrict__ We2f, const float* __restrict__ be2,
    const int* __restrict__ src, const int* __restrict__ dst,
    bf16* __restrict__ e_fg) {
  __shared__ __align__(16) bf16 xs[16 * CIN];   // 36 KB
  __shared__ __align__(16) bf16 hs[16 * HH];    // 32 KB
  const int tid = threadIdx.x, wave = tid >> 5, lane = tid & 31;
  const int half = lane >> 4, col = lane & 15;
  const int e0 = blockIdx.x * 16;

  // stage gathered input rows as bf16
#pragma unroll
  for (int rr = 0; rr < 2; ++rr) {
    int row = wave * 2 + rr;
    int e = e0 + row;
    int s = src[e], d = dst[e];
    const float* ps = n_f + (size_t)s * DD;
    const float* pd = n_f + (size_t)d * DD;
    const float* pe = s_f + (size_t)e * SS;
    bf16* xr = xs + row * CIN;
    for (int c = lane; c < DD; c += 32) xr[c] = (bf16)ps[c];
    for (int c = lane; c < SS; c += 32) xr[DD + c] = (bf16)pe[c];
    for (int c = lane; c < DD; c += 32) xr[DD + SS + c] = (bf16)pd[c];
  }
  __syncthreads();

  // layer 1: [16 x 1152] @ [1152 x 1024]; 64 N-tiles -> 8 per wave (2 groups of 4)
  const int KT1 = CIN / 32;  // 36
#pragma unroll
  for (int g = 0; g < 2; ++g) {
    int ntbase = wave * 8 + g * 4;
    v8f acc[4];
#pragma unroll
    for (int u = 0; u < 4; ++u) {
      float bias = be1[(ntbase + u) * 16 + col];
#pragma unroll
      for (int r = 0; r < 8; ++r) acc[u][r] = bias;
    }
    gemm_panel4<KT1>(xs, CIN, lane, We1f, ntbase, acc);
#pragma unroll
    for (int u = 0; u < 4; ++u) {
      int n = (ntbase + u) * 16 + col;
#pragma unroll
      for (int r = 0; r < 8; ++r) {
        float v = acc[u][r];
        v = v > 0.f ? v : 0.f;
        int m = r + half * 8;                // C/D: M = vgpr + 8*half (ISA 7.12.2)
        hs[m * HH + n] = (bf16)v;
      }
    }
  }
  __syncthreads();

  // layer 2: [16 x 1024] @ [1024 x 512]; 32 N-tiles -> 4 per wave (1 group of 4)
  const int KT2 = HH / 32;  // 32
  {
    int ntbase = wave * 4;
    v8f acc[4];
#pragma unroll
    for (int u = 0; u < 4; ++u) {
      float bias = be2[(ntbase + u) * 16 + col];
#pragma unroll
      for (int r = 0; r < 8; ++r) acc[u][r] = bias;
    }
    gemm_panel4<KT2>(hs, HH, lane, We2f, ntbase, acc);
#pragma unroll
    for (int u = 0; u < 4; ++u) {
      int n = (ntbase + u) * 16 + col;
#pragma unroll
      for (int r = 0; r < 8; ++r) {
        int m = r + half * 8;
        e_fg[(size_t)(e0 + m) * DD + n] = (bf16)acc[u][r];
      }
    }
  }
}

// ---------------- attention logit: a[e] = e_f[e] . Wa + ba ----------------
__global__ __launch_bounds__(256) void k_att(const bf16* __restrict__ e_fg,
                                             const float* __restrict__ Wa,
                                             const float* __restrict__ ba,
                                             float* __restrict__ a) {
  int wave = threadIdx.x >> 5, lane = threadIdx.x & 31;
  int e = blockIdx.x * 8 + wave;
  if (e >= EE) return;
  const bf16* row = e_fg + (size_t)e * DD;
  float s = 0.f;
  for (int c = lane; c < DD; c += 32) s += (float)row[c] * Wa[c];
#pragma unroll
  for (int off = 16; off; off >>= 1) s += __shfl_xor(s, off, 32);
  if (lane == 0) a[e] = s + ba[0];
}

// ---------------- segment softmax helpers ----------------
__global__ __launch_bounds__(256) void k_init(unsigned* __restrict__ m_ord,
                                              float* __restrict__ den,
                                              float* __restrict__ z_f) {
  int i = blockIdx.x * 256 + threadIdx.x;
  if (i < NN) { m_ord[i] = 0u; den[i] = 0.f; }   // 0 == below enc(any finite/-inf)
  if (i < NN * DD) z_f[i] = 0.f;
}

__global__ __launch_bounds__(256) void k_segmax(const float* __restrict__ a,
                                                const int* __restrict__ dst,
                                                unsigned* __restrict__ m_ord) {
  int e = blockIdx.x * 256 + threadIdx.x;
  if (e >= EE) return;
  atomicMax(&m_ord[dst[e]], enc_f32(a[e]));
}

__global__ __launch_bounds__(256) void k_segexp(const float* __restrict__ a,
                                                const int* __restrict__ dst,
                                                const unsigned* __restrict__ m_ord,
                                                float* __restrict__ ex,
                                                float* __restrict__ den) {
  int e = blockIdx.x * 256 + threadIdx.x;
  if (e >= EE) return;
  int d = dst[e];
  float v = expf(a[e] - dec_f32(m_ord[d]));
  ex[e] = v;
  atomicAdd(&den[d], v);
}

__global__ __launch_bounds__(256) void k_alpha(const float* __restrict__ ex,
                                               const int* __restrict__ dst,
                                               const float* __restrict__ den,
                                               float* __restrict__ alpha) {
  int e = blockIdx.x * 256 + threadIdx.x;
  if (e >= EE) return;
  alpha[e] = ex[e] / den[dst[e]];
}

// ---------------- z_f scatter: z_f[dst] += alpha*(n_f[src]+e_f) ----------------
__global__ __launch_bounds__(256) void k_scatter(const float* __restrict__ n_f,
                                                 const bf16* __restrict__ e_fg,
                                                 const float* __restrict__ alpha,
                                                 const int* __restrict__ src,
                                                 const int* __restrict__ dst,
                                                 float* __restrict__ z_f) {
  int wave = threadIdx.x >> 5, lane = threadIdx.x & 31;
  int e = blockIdx.x * 8 + wave;
  if (e >= EE) return;
  float al = alpha[e];
  int s = src[e], d = dst[e];
  const float* ps = n_f + (size_t)s * DD;
  const bf16* pe = e_fg + (size_t)e * DD;
  float* pz = z_f + (size_t)d * DD;
  for (int c = lane; c < DD; c += 32)
    atomicAdd(&pz[c], al * (ps[c] + (float)pe[c]));
}

// ---------------- node update: new_n = [n_f | z_f] @ Wn + bn ----------------
__global__ __launch_bounds__(256) void k_node_mlp(const float* __restrict__ n_f,
                                                  const float* __restrict__ z_f,
                                                  const bf16* __restrict__ Wnf,
                                                  const float* __restrict__ bn,
                                                  float* __restrict__ new_n) {
  __shared__ __align__(16) bf16 xs[16 * HH];  // 16 x 1024
  const int tid = threadIdx.x, wave = tid >> 5, lane = tid & 31;
  const int half = lane >> 4, col = lane & 15;
  const int node0 = blockIdx.x * 16;
#pragma unroll
  for (int rr = 0; rr < 2; ++rr) {
    int row = wave * 2 + rr;
    int node = node0 + row;
    const float* p1 = n_f + (size_t)node * DD;
    const float* p2 = z_f + (size_t)node * DD;
    bf16* xr = xs + row * HH;
    for (int c = lane; c < DD; c += 32) xr[c] = (bf16)p1[c];
    for (int c = lane; c < DD; c += 32) xr[DD + c] = (bf16)p2[c];
  }
  __syncthreads();
  const int KT = HH / 32;  // 32
  {
    int ntbase = wave * 4;
    v8f acc[4];
#pragma unroll
    for (int u = 0; u < 4; ++u) {
      float bias = bn[(ntbase + u) * 16 + col];
#pragma unroll
      for (int r = 0; r < 8; ++r) acc[u][r] = bias;
    }
    gemm_panel4<KT>(xs, HH, lane, Wnf, ntbase, acc);
#pragma unroll
    for (int u = 0; u < 4; ++u) {
      int n = (ntbase + u) * 16 + col;
#pragma unroll
      for (int r = 0; r < 8; ++r) {
        int m = r + half * 8;
        new_n[(size_t)(node0 + m) * DD + n] = acc[u][r];
      }
    }
  }
}

// ---------------- prediction head ----------------
__global__ __launch_bounds__(256) void k_pred(
    const float* __restrict__ new_n, const float* __restrict__ s_f,
    const bf16* __restrict__ Wp1f, const float* __restrict__ bp1,
    const float* __restrict__ Wp2, const float* __restrict__ bp2,
    const int* __restrict__ src, const int* __restrict__ dst,
    const int* __restrict__ pidx, float* __restrict__ out) {
  __shared__ __align__(16) bf16 xs[16 * CIN];      // 36 KB
  __shared__ __align__(16) bf16 hs[16 * DD];       // 16 KB
  const int tid = threadIdx.x, wave = tid >> 5, lane = tid & 31;
  const int half = lane >> 4, col = lane & 15;
  const int p0 = blockIdx.x * 16;
#pragma unroll
  for (int rr = 0; rr < 2; ++rr) {
    int row = wave * 2 + rr;
    int p = p0 + row;
    int e = pidx[p];
    int ps = src[e], pd = dst[e];
    const float* a1 = new_n + (size_t)ps * DD;
    const float* a2 = s_f + (size_t)e * SS;
    const float* a3 = new_n + (size_t)pd * DD;
    bf16* xr = xs + row * CIN;
    for (int c = lane; c < DD; c += 32) xr[c] = (bf16)a1[c];
    for (int c = lane; c < SS; c += 32) xr[DD + c] = (bf16)a2[c];
    for (int c = lane; c < DD; c += 32) xr[DD + SS + c] = (bf16)a3[c];
  }
  __syncthreads();

  // layer 1: [16 x 1152] @ [1152 x 512]; 32 N-tiles -> 4 per wave (1 group of 4)
  const int KT1 = CIN / 32;  // 36
  {
    int ntbase = wave * 4;
    v8f acc[4];
#pragma unroll
    for (int u = 0; u < 4; ++u) {
      float bias = bp1[(ntbase + u) * 16 + col];
#pragma unroll
      for (int r = 0; r < 8; ++r) acc[u][r] = bias;
    }
    gemm_panel4<KT1>(xs, CIN, lane, Wp1f, ntbase, acc);
#pragma unroll
    for (int u = 0; u < 4; ++u) {
      int n = (ntbase + u) * 16 + col;
#pragma unroll
      for (int r = 0; r < 8; ++r) {
        float v = acc[u][r];
        v = v > 0.f ? v : 0.f;
        int m = r + half * 8;
        hs[m * DD + n] = (bf16)v;
      }
    }
  }
  __syncthreads();

  // layer 2: [16 x 512] @ [512 x 1]
#pragma unroll
  for (int rr = 0; rr < 2; ++rr) {
    int row = wave * 2 + rr;
    float s = 0.f;
    const bf16* hr = hs + row * DD;
    for (int c = lane; c < DD; c += 32) s += (float)hr[c] * Wp2[c];
#pragma unroll
    for (int off = 16; off; off >>= 1) s += __shfl_xor(s, off, 32);
    if (lane == 0) out[p0 + row] = s + bp2[0];
  }
}

// ---------------- host launcher ----------------
extern "C" void kernel_launch(void* const* d_in, const int* in_sizes, int n_in,
                              void* d_out, int out_size, void* d_ws, size_t ws_size,
                              hipStream_t stream) {
  const float* n_f = (const float*)d_in[0];
  const float* s_f = (const float*)d_in[1];
  const float* We1 = (const float*)d_in[2];
  const float* be1 = (const float*)d_in[3];
  const float* We2 = (const float*)d_in[4];
  const float* be2 = (const float*)d_in[5];
  const float* Wa  = (const float*)d_in[6];
  const float* ba  = (const float*)d_in[7];
  const float* Wn  = (const float*)d_in[8];
  const float* bn  = (const float*)d_in[9];
  const float* Wp1 = (const float*)d_in[10];
  const float* bp1 = (const float*)d_in[11];
  const float* Wp2 = (const float*)d_in[12];
  const float* bp2 = (const float*)d_in[13];
  const int* src = (const int*)d_in[14];
  const int* dst = (const int*)d_in[15];
  const int* pidx = (const int*)d_in[16];
  float* out = (float*)d_out;

  // workspace carve-up (bytes, 256-aligned chunks)
  char* ws = (char*)d_ws;
  size_t off = 0;
  auto take = [&](size_t bytes) {
    char* p = ws + off;
    off += (bytes + 255) & ~(size_t)255;
    return p;
  };
  bf16* We1f = (bf16*)take((size_t)CIN * HH * 2);    // 2.36 MB
  bf16* We2f = (bf16*)take((size_t)HH * DD * 2);     // 1.05 MB
  bf16* Wnf  = (bf16*)take((size_t)(2 * DD) * DD * 2);
  bf16* Wp1f = (bf16*)take((size_t)CIN * DD * 2);
  bf16* e_fg = (bf16*)take((size_t)EE * DD * 2);     // 164 MB
  float* a    = (float*)take((size_t)EE * 4);
  float* ex   = (float*)take((size_t)EE * 4);
  float* alpha = (float*)take((size_t)EE * 4);
  unsigned* m_ord = (unsigned*)take((size_t)NN * 4);
  float* den  = (float*)take((size_t)NN * 4);
  float* z_f  = (float*)take((size_t)NN * DD * 4);   // 41 MB
  float* newn = (float*)take((size_t)NN * DD * 4);   // 41 MB
  (void)ws_size; (void)n_in; (void)in_sizes; (void)out_size;

  // 1) weight prep (bf16, fragment-swizzled)
  {
    int t1 = (CIN / 32) * (HH / 16) * 512;
    k_prep_weight<<<(t1 + 255) / 256, 256, 0, stream>>>(We1, We1f, CIN, HH);
    int t2 = (HH / 32) * (DD / 16) * 512;
    k_prep_weight<<<(t2 + 255) / 256, 256, 0, stream>>>(We2, We2f, HH, DD);
    int t3 = ((2 * DD) / 32) * (DD / 16) * 512;
    k_prep_weight<<<(t3 + 255) / 256, 256, 0, stream>>>(Wn, Wnf, 2 * DD, DD);
    int t4 = (CIN / 32) * (DD / 16) * 512;
    k_prep_weight<<<(t4 + 255) / 256, 256, 0, stream>>>(Wp1, Wp1f, CIN, DD);
  }

  // 2) init segment state + z_f
  k_init<<<(NN * DD + 255) / 256, 256, 0, stream>>>(m_ord, den, z_f);

  // 3) edge MLP (WMMA)
  k_edge_mlp<<<EE / 16, 256, 0, stream>>>(n_f, s_f, We1f, be1, We2f, be2, src, dst, e_fg);

  // 4) attention logits
  k_att<<<EE / 8, 256, 0, stream>>>(e_fg, Wa, ba, a);

  // 5) segment softmax
  k_segmax<<<(EE + 255) / 256, 256, 0, stream>>>(a, dst, m_ord);
  k_segexp<<<(EE + 255) / 256, 256, 0, stream>>>(a, dst, m_ord, ex, den);
  k_alpha<<<(EE + 255) / 256, 256, 0, stream>>>(ex, dst, den, alpha);

  // 6) weighted scatter to z_f
  k_scatter<<<EE / 8, 256, 0, stream>>>(n_f, e_fg, alpha, src, dst, z_f);

  // 7) node linear (WMMA)
  k_node_mlp<<<NN / 16, 256, 0, stream>>>(n_f, z_f, Wnf, bn, newn);

  // 8) prediction head (WMMA + reduce)
  k_pred<<<PP / 16, 256, 0, stream>>>(newn, s_f, Wp1f, bp1, Wp2, bp2, src, dst, pidx, out);
}
